// LSTMRecDQN_69174743269833
// MI455X (gfx1250) — matrix operand order repaired and checked by reading, not compiled, not run on hardware
//
#include <hip/hip_runtime.h>
#include <hip/hip_bf16.h>

// ---- types ----------------------------------------------------------------
typedef _Float16 h16;
typedef __attribute__((ext_vector_type(16))) _Float16 v16h;
typedef __attribute__((ext_vector_type(8)))  float    v8f;
typedef __attribute__((ext_vector_type(2)))  __fp16   v2fp;

#define DEVINL static __device__ __forceinline__

union Frag16 { unsigned u[8]; uint4 q[2]; v16h v; };

// ---- small helpers --------------------------------------------------------
DEVINL unsigned swz16u(unsigned x) {
  // ds_swizzle group-of-32: and=0x1f, or=0, xor=0x10 -> SWAPX16 (lane ^ 16)
  return (unsigned)__builtin_amdgcn_ds_swizzle((int)x, 0x401F);
}
DEVINL int swz16i(int x) { return __builtin_amdgcn_ds_swizzle(x, 0x401F); }

DEVINL unsigned pk2(float a, float b) {
  union { v2fp h; unsigned u; } c;
  c.h = __builtin_amdgcn_cvt_pkrtz(a, b);
  return c.u;
}

DEVINL float tanh_f(float x) {
  float r;
  asm("v_tanh_f32 %0, %1" : "=v"(r) : "v"(x));
  return r;
}
DEVINL float sigm(float x) { return __builtin_fmaf(0.5f, tanh_f(0.5f * x), 0.5f); }

DEVINL v8f wmma(v16h a, v16h b, v8f c) {
  return __builtin_amdgcn_wmma_f32_16x16x32_f16(false, a, false, b, (short)0, c,
                                                false, false);
}

// A-operand fragment: lane covers matrix row n; element e -> K=chunk_base+16*lane_half+e
// (caller passes k0 = chunk_base + 8*lane_half).
DEVINL v16h load_afrag(const float* __restrict__ W, int stride, int n, int k0,
                       int kvalid, bool rowvalid) {
  Frag16 f;
#pragma unroll
  for (int p = 0; p < 8; ++p) {
    int e0 = 2 * p;
    int ka = k0 + (k0 & 8) + e0;
    int kb = ka + 1;
    float va = (rowvalid && ka < kvalid) ? W[n * stride + ka] : 0.0f;
    float vb = (rowvalid && kb < kvalid) ? W[n * stride + kb] : 0.0f;
    f.u[p] = pk2(va, vb);
  }
  return f.v;
}

// Same, but K slot 7 carries a folded bias (the matching B operand has a
// constant 1.0 planted in feature slot 7).
DEVINL v16h load_afrag_bias(const float* __restrict__ W, int stride, int n,
                            int kvalid, float bias, int half) {
  Frag16 f;
#pragma unroll
  for (int p = 0; p < 8; ++p) {
    int e0 = 2 * p;
    int ka = 8 * half + (e0 < 8 ? e0 : e0 + 8);
    int kb = ka + 1;
    float va = (ka < kvalid) ? W[n * stride + ka] : (ka == 7 ? bias : 0.0f);
    float vb = (kb < kvalid) ? W[n * stride + kb] : (kb == 7 ? bias : 0.0f);
    f.u[p] = pk2(va, vb);
  }
  return f.v;
}

// Build the two B-operand chunks (K=0..63) of hT (64 x 16) from per-lane
// packed h values: lane holds k in {16j + r + 8*half}. Only a lane<->lane+16
// exchange (SWAPX16) is needed.
DEVINL void mk_bh64(const unsigned (&hp)[4][4], int half, v16h (&out)[2]) {
#pragma unroll
  for (int c = 0; c < 2; ++c) {
    Frag16 f;
#pragma unroll
    for (int p = 0; p < 4; ++p) {
      unsigned lo = hp[2 * c][p];      // k = 32c + 2p   + 8*half
      unsigned hi = hp[2 * c + 1][p];  // k = 32c + 16 + 2p + 8*half
      unsigned send = half ? lo : hi;
      unsigned rec  = swz16u(send);
      f.u[p]     = half ? rec : lo;   // K = 32c + 16*half + 0..7
      f.u[4 + p] = half ? hi  : rec;  // K = 32c + 16*half + 8..15
    }
    out[c] = f.v;
  }
}

DEVINL void pack64(const float (&x)[4][8], unsigned (&q)[4][4]) {
#pragma unroll
  for (int j = 0; j < 4; ++j)
#pragma unroll
    for (int p = 0; p < 4; ++p) q[j][p] = pk2(x[j][2 * p], x[j][2 * p + 1]);
}

// Activation for one hidden sub-block jj given its 4 gate tiles (i,f,g,o).
DEVINL void lstm_group(const v8f (&g4)[4], float (&cstj)[8], unsigned (&hpj)[4],
                       float (&capj)[8], bool take) {
  float hv[8];
#pragma unroll
  for (int r = 0; r < 8; ++r) {
    float iv = g4[0][r];
    float fv = g4[1][r];
    float gv = g4[2][r];
    float ov = g4[3][r];
    float cn = sigm(fv) * cstj[r] + sigm(iv) * tanh_f(gv);
    cstj[r] = cn;
    float hh = sigm(ov) * tanh_f(cn);
    hv[r] = hh;
    capj[r] = take ? hh : capj[r];
  }
#pragma unroll
  for (int p = 0; p < 4; ++p) hpj[p] = pk2(hv[2 * p], hv[2 * p + 1]);
}

// ===========================================================================
// Kernel 1: one inner LSTM branch. One wave per 16-row batch tile.
// slot o: 0 -> x2 (s_hist[:,0], weights set 2), 1 -> x1, 2 -> x0 (s).
// Writes x feature vector (128 f16) per (row, slot) to workspace.
// ===========================================================================
__global__ __launch_bounds__(32) void inner_k(
    const float* __restrict__ s, const float* __restrict__ s_hist,
    const int* __restrict__ hist_len,
    const float* __restrict__ WOS, const float* __restrict__ bOS,
    const float* __restrict__ Wih, const float* __restrict__ Whh,
    const float* __restrict__ bih, const float* __restrict__ bhh,
    const float* __restrict__ WTS, const float* __restrict__ bTS,
    h16* __restrict__ xws, int o) {
  __shared__ h16 sts[16 * 64 * 8 + 8];  // [row][t][8] f16 feats (+1 @7); +zslot
  __shared__ h16 sos[16 * 8];           // [row][8]    f16 OS feats (+1 @7)
  __shared__ h16 wih_lds[16 * 32 * 16]; // [tile][lane][16] full 32B A frags

  const int lane = threadIdx.x;
  const int half = lane >> 4;
  const int row  = lane & 15;
  const int b    = blockIdx.x * 16 + row;
  const int hl   = hist_len[b];
  const uint4 z4 = make_uint4(0, 0, 0, 0);

  const float* src;
  bool zerosrc = false;
  if (o == 2) {
    src = s + (long)b * 391;
  } else if (o == 0) {
    src = s_hist + (long)b * 2 * 391;
    zerosrc = hl < 2;
  } else {
    int i1 = hl - 1; if (i1 < 0) i1 = 0;
    src = s_hist + ((long)b * 2 + i1) * 391;
    zerosrc = hl < 2;
  }

  // ---- stage inputs (lane half 0: steps 0..31, half 1: steps 32..63) ------
  int cnt = 0;
  const int t0 = half * 32;
  for (int tt = 0; tt < 32; ++tt) {
    int t = t0 + tt;
    const float* sp = src + 7 + t * 6;
    float f[6];
#pragma unroll
    for (int j = 0; j < 6; ++j) {
      float v = sp[j];
      f[j] = zerosrc ? 0.0f : v;
    }
    cnt += (f[0] == f[0]) ? 1 : 0;  // valid iff feature0 not NaN (post zero-mask)
    h16* dst = &sts[(row * 64 + t) * 8];
#pragma unroll
    for (int j = 0; j < 6; ++j) {
      float v = f[j];
      v = (v == v) ? v : 0.0f;
      dst[j] = (h16)v;
    }
    dst[6] = (h16)0.0f;
    dst[7] = (h16)1.0f;  // bias lane: pairs with K=7 column of Wih fragment
  }
  if (half == 0) {
#pragma unroll
    for (int j = 0; j < 7; ++j) {
      float v = src[j];
      v = zerosrc ? 0.0f : v;
      v = (v == v) ? v : 0.0f;
      sos[row * 8 + j] = (h16)v;
    }
    sos[row * 8 + 7] = (h16)1.0f;  // bias lane for WOS
  }
  if (lane == 16) {  // zero slot read by lanes >= 16 in the step loop
    *(uint4*)&sts[16 * 64 * 8] = z4;
  }
  // ---- Wih(+bias) full A fragments into LDS (32B per tile per lane) -------
#pragma unroll
  for (int j = 0; j < 16; ++j) {
    int n = 16 * j + row;
    float bias = bih[n] + bhh[n];
    Frag16 t;
    t.v = load_afrag_bias(Wih, 6, n, 6, bias, 0);
    uint4 q0 = half ? z4 : t.q[0];  // lanes >=16 hold zeros (K=8..15,24..31)
    uint4* dst = (uint4*)&wih_lds[(j * 32 + lane) * 16];
    dst[0] = q0;
    dst[1] = z4;  // K=16..23 / 24..31 columns are zero
  }
  int nvalid = cnt + swz16i(cnt);
  int hidx = nvalid - 1;
  if (hidx < 0) hidx = 0;
  __syncthreads();

  // ---- resident Whh A fragments -------------------------------------------
  v16h whhA[16][2];
#pragma unroll
  for (int j = 0; j < 16; ++j) {
    int nrow = 16 * j + row;
    whhA[j][0] = load_afrag(Whh, 64, nrow, 8 * half, 64, true);
    whhA[j][1] = load_afrag(Whh, 64, nrow, 32 + 8 * half, 64, true);
  }

  float    cst[4][8] = {};
  unsigned hp[4][4]  = {};
  float    xts[4][8] = {};

  // per-lane bx addressing: lanes<16 step through their row, lanes>=16 read
  // the zero slot every step (no data selects needed).
  const int xoff0 = half ? (16 * 64 * 8) : (row * 512);
  const int xstr  = half ? 0 : 8;

  // ---- 64 recurrent steps: 48 v_wmma per step -----------------------------
#pragma unroll 1
  for (int t = 0; t < 64; ++t) {
    // B operand: xT step tile (K = 6 feats + 1.0 bias lane, cols = batch)
    uint4 xq = *(const uint4*)&sts[xoff0 + t * xstr];
    Frag16 bx;
    bx.q[0] = xq;
    bx.q[1] = z4;
    // B operand: hT (two K chunks) from packed h via SWAPX16
    v16h bh[2];
    mk_bh64(hp, half, bh);
    bool take = (t == hidx);

    v8f z8 = {};
#pragma unroll
    for (int jj = 0; jj < 4; ++jj) {
      v8f g4[4];
#pragma unroll
      for (int g = 0; g < 4; ++g) {
        int j = 4 * g + jj;
        Frag16 wa;
        const uint4* wp = (const uint4*)&wih_lds[(j * 32 + lane) * 16];
        wa.q[0] = wp[0];
        wa.q[1] = wp[1];
        v8f a = wmma(wa.v, bx.v, z8);      // x part + bias (K slot 7)
        a = wmma(whhA[j][0], bh[0], a);    // h part, K 0..31
        a = wmma(whhA[j][1], bh[1], a);    // h part, K 32..63
        g4[g] = a;
      }
      lstm_group(g4, cst[jj], hp[jj], xts[jj], take);
    }
  }

  // ---- xTS = relu(WTS * xtsT + bTS) ---------------------------------------
  unsigned xp[4][4];
  pack64(xts, xp);
  v16h bxt[2];
  mk_bh64(xp, half, bxt);

  float xout[8][8];
#pragma unroll
  for (int j = 0; j < 4; ++j) {
    int nrow = 16 * j + row;
    v16h a0 = load_afrag(WTS, 64, nrow, 8 * half, 64, true);
    v16h a1 = load_afrag(WTS, 64, nrow, 32 + 8 * half, 64, true);
    v8f bc;
#pragma unroll
    for (int r = 0; r < 8; ++r) bc[r] = bTS[16 * j + r + 8 * half];
    v8f a = wmma(a0, bxt[0], bc);
    a = wmma(a1, bxt[1], a);
#pragma unroll
    for (int r = 0; r < 8; ++r) xout[4 + j][r] = fmaxf(a[r], 0.0f);
  }

  // ---- xOS = relu(WOS * sOST + bOS), bias folded at K=7 -------------------
  uint4 oq = *(const uint4*)&sos[row * 8];
  Frag16 bo;
  bo.q[0] = half ? z4 : oq;
  bo.q[1] = z4;
  v8f z8b = {};
#pragma unroll
  for (int j = 0; j < 4; ++j) {
    int nrow = 16 * j + row;
    v16h a0 = load_afrag_bias(WOS, 7, nrow, 7, bOS[nrow], half);
    v8f a = wmma(a0, bo.v, z8b);
#pragma unroll
    for (int r = 0; r < 8; ++r) xout[j][r] = fmaxf(a[r], 0.0f);
  }

  // ---- store 128 f16 features: lane writes feats 16j+8*half+0..7 ----------
#pragma unroll
  for (int j = 0; j < 8; ++j) {
    uint4 q;
    q.x = pk2(xout[j][0], xout[j][1]);
    q.y = pk2(xout[j][2], xout[j][3]);
    q.z = pk2(xout[j][4], xout[j][5]);
    q.w = pk2(xout[j][6], xout[j][7]);
    h16* dst = xws + ((long)b * 3 + o) * 128 + 16 * j + 8 * half;
    *(uint4*)dst = q;
  }
}

// ===========================================================================
// Kernel 2: history roll + outer LSTM (3 steps) + MLP head. One wave / 16 rows.
// ===========================================================================
__global__ __launch_bounds__(32) void outer_k(
    const h16* __restrict__ xws, const int* __restrict__ hist_len,
    const float* __restrict__ Wiho, const float* __restrict__ Whho,
    const float* __restrict__ biho, const float* __restrict__ bhho,
    const float* __restrict__ W1, const float* __restrict__ b1,
    const float* __restrict__ W2, const float* __restrict__ b2,
    float* __restrict__ out) {
  const int lane = threadIdx.x;
  const int half = lane >> 4;
  const int row  = lane & 15;
  const int b    = blockIdx.x * 16 + row;
  const int hl   = hist_len[b];

  v8f biasC[16];
#pragma unroll
  for (int j = 0; j < 16; ++j)
#pragma unroll
    for (int r = 0; r < 8; ++r) {
      int g = 16 * j + r + 8 * half;
      biasC[j][r] = biho[g] + bhho[g];
    }

  float    cst[4][8] = {};
  unsigned hp[4][4]  = {};
  float    hid[4][8] = {};

#pragma unroll 1
  for (int t = 0; t < 3; ++t) {
    // rolled slot: step t consumes orig[(t + 2 - hl) mod 3], orig = [x2,x1,x0]
    int osel = t + 2 - hl;
    if (osel >= 3) osel -= 3;
    const h16* xb = xws + ((long)b * 3 + osel) * 128;

    // B operand: x_selT, 4 K-chunks; lane covers feats 32c+16*half+0..15
    v16h bx[4];
#pragma unroll
    for (int c = 0; c < 4; ++c) {
      const uint4* p = (const uint4*)(xb + 32 * c + 16 * half);
      Frag16 f;
      f.q[0] = p[0];
      f.q[1] = p[1];
      bx[c] = f.v;
    }
    v16h bh[2];
    mk_bh64(hp, half, bh);
    bool take = (t == hl);

#pragma unroll
    for (int jj = 0; jj < 4; ++jj) {
      v8f g4[4];
#pragma unroll
      for (int g = 0; g < 4; ++g) {
        int j = 4 * g + jj;
        int nrow = 16 * j + row;
        v8f a = biasC[j];
#pragma unroll
        for (int c = 0; c < 4; ++c) {
          v16h af = load_afrag(Wiho, 128, nrow, 32 * c + 8 * half, 128, true);
          a = wmma(af, bx[c], a);
        }
#pragma unroll
        for (int c = 0; c < 2; ++c) {
          v16h af = load_afrag(Whho, 64, nrow, 32 * c + 8 * half, 64, true);
          a = wmma(af, bh[c], a);
        }
        g4[g] = a;
      }
      lstm_group(g4, cst[jj], hp[jj], hid[jj], take);
    }
  }

  // ---- x = relu(W1 * hiddenT + b1) ----------------------------------------
  unsigned hq[4][4];
  pack64(hid, hq);
  v16h bh1[2];
  mk_bh64(hq, half, bh1);

  float xh[4][8];
#pragma unroll
  for (int j = 0; j < 4; ++j) {
    int nrow = 16 * j + row;
    v16h a0 = load_afrag(W1, 64, nrow, 8 * half, 64, true);
    v16h a1 = load_afrag(W1, 64, nrow, 32 + 8 * half, 64, true);
    v8f bc;
#pragma unroll
    for (int r = 0; r < 8; ++r) bc[r] = b1[16 * j + r + 8 * half];
    v8f a = wmma(a0, bh1[0], bc);
    a = wmma(a1, bh1[1], a);
#pragma unroll
    for (int r = 0; r < 8; ++r) xh[j][r] = fmaxf(a[r], 0.0f);
  }

  // ---- out = W2 * xT + b2 (9 outputs, padded to one 16x16 tile) -----------
  unsigned xq2[4][4];
  pack64(xh, xq2);
  v16h bx2[2];
  mk_bh64(xq2, half, bx2);

  bool rv = row < 9;
  v16h w20 = load_afrag(W2, 64, row, 8 * half, 64, rv);
  v16h w21 = load_afrag(W2, 64, row, 32 + 8 * half, 64, rv);
  v8f bc2;
#pragma unroll
  for (int r = 0; r < 8; ++r) {
    int nn = r + 8 * half;
    bc2[r] = (nn < 9) ? b2[nn] : 0.0f;
  }
  v8f a = wmma(w20, bx2[0], bc2);
  a = wmma(w21, bx2[1], a);

#pragma unroll
  for (int r = 0; r < 8; ++r) {
    int nn = r + 8 * half;
    if (nn < 9) out[(long)b * 9 + nn] = a[r];
  }
}

// ===========================================================================
extern "C" void kernel_launch(void* const* d_in, const int* in_sizes, int n_in,
                              void* d_out, int out_size, void* d_ws,
                              size_t ws_size, hipStream_t stream) {
  (void)in_sizes; (void)n_in; (void)out_size; (void)ws_size;
  const float* s        = (const float*)d_in[0];
  const float* s_hist   = (const float*)d_in[1];
  // d_in[2] = a_hist (unused by the reference)
  const int*   hist_len = (const int*)d_in[3];
  auto W = [&](int i) { return (const float*)d_in[i]; };

  h16*   xws = (h16*)d_ws;  // 8192 * 3 * 128 f16 = 6 MB
  float* out = (float*)d_out;

  // slot o=0 -> x2 (weights set 2 @ d_in[20..27]); o=1 -> set1; o=2 -> set0
  const int base[3] = {20, 12, 4};
  for (int o = 0; o < 3; ++o) {
    int bb = base[o];
    inner_k<<<512, 32, 0, stream>>>(s, s_hist, hist_len, W(bb + 0), W(bb + 1),
                                    W(bb + 2), W(bb + 3), W(bb + 4), W(bb + 5),
                                    W(bb + 6), W(bb + 7), xws, o);
  }
  outer_k<<<512, 32, 0, stream>>>(xws, hist_len, W(28), W(29), W(30), W(31),
                                  W(32), W(33), W(34), W(35), out);
}
